// mLSTMBlock_22479858827428
// MI455X (gfx1250) — compile-verified, matrix operand order checked by reading, not run on hardware
//
#include <hip/hip_runtime.h>
#include <hip/hip_bf16.h>

// ---------------- problem dims ----------------
#define BB   8192
#define DD   1024
#define NHH  16
#define HSS  64
#define HHH  1024
#define UPF  2048
#define EPSF 1e-5f

// ---------------- WMMA types ----------------
typedef __attribute__((ext_vector_type(16))) __bf16 bf16x16;
typedef __attribute__((ext_vector_type(8)))  float  f32x8;

__device__ __forceinline__ unsigned short f2bf(float f) {
  union { float f; unsigned u; } v; v.f = f;
  unsigned r = v.u + 0x7FFFu + ((v.u >> 16) & 1u);   // round-to-nearest-even
  return (unsigned short)(r >> 16);
}
__device__ __forceinline__ float sigmoidf_(float x) { return 1.0f / (1.0f + __expf(-x)); }
__device__ __forceinline__ float siluf_(float x)    { return x * sigmoidf_(x); }

// ---- CDNA5 async global->LDS copy (ASYNCcnt path, bypasses VGPRs) ----
__device__ __forceinline__ void async_copy_b128(void* lds_dst, const void* gsrc) {
  unsigned ldsoff = (unsigned)(uintptr_t)lds_dst;              // low 32 bits = LDS offset
  unsigned long long ga = (unsigned long long)gsrc;
  asm volatile("global_load_async_to_lds_b128 %0, %1, off"
               :: "v"(ldsoff), "v"(ga) : "memory");
}
__device__ __forceinline__ void wait_async0() {
  asm volatile("s_wait_asynccnt 0" ::: "memory");
}

// ---------------- weight convert + transpose: W[K,N] f32 -> Wt[N,K] bf16 ----------------
__global__ void convert_transpose(const float* __restrict__ W,
                                  unsigned short* __restrict__ Wt,
                                  int K, int N) {
  int id = blockIdx.x * 256 + threadIdx.x;
  if (id >= K * N) return;
  int k = id / N, n = id % N;
  Wt[(size_t)n * K + k] = f2bf(W[id]);
}

// ---------------- layernorm -> bf16 ----------------
__global__ __launch_bounds__(256)
void layernorm_bf16(const float* __restrict__ x, const float* __restrict__ w,
                    const float* __restrict__ b, unsigned short* __restrict__ out) {
  __shared__ float s1[256], s2[256];
  int row = blockIdx.x, t = threadIdx.x;
  const float* xr = x + (size_t)row * DD;
  float v[4], s = 0.f, q = 0.f;
#pragma unroll
  for (int e = 0; e < 4; e++) { v[e] = xr[t + e * 256]; s += v[e]; q += v[e] * v[e]; }
  s1[t] = s; s2[t] = q; __syncthreads();
  for (int st = 128; st > 0; st >>= 1) {
    if (t < st) { s1[t] += s1[t + st]; s2[t] += s2[t + st]; }
    __syncthreads();
  }
  float mu = s1[0] * (1.0f / DD);
  float var = s2[0] * (1.0f / DD) - mu * mu;
  float rs = rsqrtf(var + EPSF);
#pragma unroll
  for (int e = 0; e < 4; e++) {
    int c = t + e * 256;
    out[(size_t)row * DD + c] = f2bf((v[e] - mu) * rs * w[c] + b[c]);
  }
}

// ---------------- generic bf16 WMMA GEMM, async double-buffered ----------------
// C[M,N] = A[M,K](bf16) @ Bt[N,K](bf16)^T [+ bias] [+ resid]; outputs f32 / bf16
#define TM 128
#define TN 128
#define TK 32
#define LSTR 40   // LDS row stride in shorts (32 data + 8 pad) -> 80B, 16B aligned

template <bool HB, bool HR, bool WF, bool WBF>
__global__ __launch_bounds__(256, 2)
void gemm_bf16(const unsigned short* __restrict__ A,
               const unsigned short* __restrict__ Bt,
               const float* __restrict__ bias,
               const float* __restrict__ resid,
               float* __restrict__ Cf,
               unsigned short* __restrict__ Cbf,
               int M, int N, int K) {
  __shared__ __align__(16) unsigned short lA[2][TM * LSTR];
  __shared__ __align__(16) unsigned short lB[2][TN * LSTR];

  const int tid  = threadIdx.x;
  const int wave = tid >> 5, lane = tid & 31;
  const int wm = wave >> 2;      // 0..1  (64-row wave tile)
  const int wn = wave & 3;       // 0..3  (32-col wave tile)
  const int lrow = tid >> 2;             // staging: row 0..63 (+64)
  const int lcol = (tid & 3) << 3;       // staging: chunk of 8 shorts
  const int rowA0 = blockIdx.y * TM;
  const int rowB0 = blockIdx.x * TN;
  const int hsel = lane >> 4;            // K-half select per ISA A/B layout
  const int mr   = lane & 15;
  const int kc   = hsel << 3;

  // per-thread staging source pointers (advance by k0 each step)
  const unsigned short* gA0 = &A[(size_t)(rowA0 + lrow) * K + lcol];
  const unsigned short* gA1 = &A[(size_t)(rowA0 + lrow + 64) * K + lcol];
  const unsigned short* gB0 = &Bt[(size_t)(rowB0 + lrow) * K + lcol];
  const unsigned short* gB1 = &Bt[(size_t)(rowB0 + lrow + 64) * K + lcol];

  f32x8 acc[4][2];
#pragma unroll
  for (int i = 0; i < 4; i++)
#pragma unroll
    for (int j = 0; j < 2; j++)
#pragma unroll
      for (int g = 0; g < 8; g++) acc[i][j][g] = 0.0f;

  // prologue: async-stage first K tile into buffer 0
  async_copy_b128(&lA[0][lrow * LSTR + lcol],        gA0);
  async_copy_b128(&lA[0][(lrow + 64) * LSTR + lcol], gA1);
  async_copy_b128(&lB[0][lrow * LSTR + lcol],        gB0);
  async_copy_b128(&lB[0][(lrow + 64) * LSTR + lcol], gB1);
  wait_async0();
  __syncthreads();

  int buf = 0;
  for (int k0 = 0; k0 < K; k0 += TK) {
    // async-stage next K tile into the other buffer while computing on this one
    if (k0 + TK < K) {
      int kn = k0 + TK;
      async_copy_b128(&lA[buf ^ 1][lrow * LSTR + lcol],        gA0 + kn);
      async_copy_b128(&lA[buf ^ 1][(lrow + 64) * LSTR + lcol], gA1 + kn);
      async_copy_b128(&lB[buf ^ 1][lrow * LSTR + lcol],        gB0 + kn);
      async_copy_b128(&lB[buf ^ 1][(lrow + 64) * LSTR + lcol], gB1 + kn);
    }

    // per-lane fragments: two contiguous 8-bf16 chunks (K=kc..kc+7 and +16)
    bf16x16 af[4], bfr[2];
#pragma unroll
    for (int j = 0; j < 2; j++) {
      const __bf16* p = (const __bf16*)&lB[buf][(wn * 32 + j * 16 + mr) * LSTR + kc];
#pragma unroll
      for (int e = 0; e < 8; e++) { bfr[j][e] = p[e]; bfr[j][8 + e] = p[16 + e]; }
    }
#pragma unroll
    for (int i = 0; i < 4; i++) {
      const __bf16* p = (const __bf16*)&lA[buf][(wm * 64 + i * 16 + mr) * LSTR + kc];
#pragma unroll
      for (int e = 0; e < 8; e++) { af[i][e] = p[e]; af[i][8 + e] = p[16 + e]; }
    }

#pragma unroll
    for (int i = 0; i < 4; i++)
#pragma unroll
      for (int j = 0; j < 2; j++)
        acc[i][j] = __builtin_amdgcn_wmma_f32_16x16x32_bf16(
            false, af[i], false, bfr[j], (short)0, acc[i][j], false, false);

    // all waves: done reading buf, own async writes to buf^1 complete, then sync
    wait_async0();
    __syncthreads();
    buf ^= 1;
  }

  // epilogue: C element (m = i*16 + hsel*8 + g, n = mr) per ISA C/D layout
#pragma unroll
  for (int i = 0; i < 4; i++) {
#pragma unroll
    for (int j = 0; j < 2; j++) {
      int n = rowB0 + wn * 32 + j * 16 + mr;
      float bv = HB ? bias[n] : 0.0f;
#pragma unroll
      for (int g = 0; g < 8; g++) {
        int m = rowA0 + wm * 64 + i * 16 + hsel * 8 + g;
        size_t off = (size_t)m * N + n;
        float v = acc[i][j][g] + bv;
        if (HR) v += resid[off];
        if (WF)  Cf[off]  = v;
        if (WBF) Cbf[off] = f2bf(v);
      }
    }
  }
}

// ---------------- causal conv (k=4, left pad 3) + SiLU -> bf16 ----------------
__global__ void conv_silu_bf16(const float* __restrict__ xl,
                               const float* __restrict__ cw,
                               const float* __restrict__ cb,
                               unsigned short* __restrict__ out) {
  int id = blockIdx.x * 256 + threadIdx.x;   // over B*UP
  int bI = id / UPF, f = id % UPF;
  const float* r = xl + (size_t)bI * UPF;
  float a = cb[0];
#pragma unroll
  for (int t = 0; t < 4; t++) { int s = f - 3 + t; if (s >= 0) a += cw[t] * r[s]; }
  out[id] = f2bf(siluf_(a));
}

// ---------------- mLSTM gate math; also emit n^T and q^T in bf16 ----------------
__global__ void gate_kernel(const float* __restrict__ it_, const float* __restrict__ ft_,
                            const float* __restrict__ kf, const float* __restrict__ vf,
                            const float* __restrict__ qf,
                            const float* __restrict__ cprev, const float* __restrict__ nprev,
                            const float* __restrict__ mprev,
                            float* __restrict__ out_c, float* __restrict__ out_n,
                            float* __restrict__ out_m,
                            unsigned short* __restrict__ nT, unsigned short* __restrict__ qT) {
  int id = blockIdx.x * 256 + threadIdx.x;   // over B*H
  int bI = id / HHH, h = id % HHH;
  float it = it_[id], ftv = ft_[id], mp = mprev[id];
  float mt = fmaxf(ftv + mp, it);
  float ig = __expf(it - mt);
  float fg = __expf(ftv + mp - mt);
  float kk = kf[id] * 0.125f;                 // 1/sqrt(HS)
  float ct = fg * cprev[id] + ig * (vf[id] * kk);
  float nt = fg * nprev[id] + ig * kk;
  out_m[id] = mt; out_c[id] = ct; out_n[id] = nt;
  nT[(size_t)h * BB + bI] = f2bf(nt);
  qT[(size_t)h * BB + bI] = f2bf(qf[id]);
}

// ---------------- denom[m] = max_j |G[m][j]| ----------------
__global__ __launch_bounds__(256)
void rowmax_abs(const float* __restrict__ G, float* __restrict__ denom) {
  __shared__ float s[256];
  int m = blockIdx.x, t = threadIdx.x;
  const float* r = G + (size_t)m * HHH;
  float mx = 0.0f;
  for (int c = t; c < HHH; c += 256) mx = fmaxf(mx, fabsf(r[c]));
  s[t] = mx; __syncthreads();
  for (int st = 128; st > 0; st >>= 1) {
    if (t < st) s[t] = fmaxf(s[t], s[t + st]);
    __syncthreads();
  }
  if (t == 0) denom[m] = s[0];
}

// ---------------- h_t + groupnorm + skip + silu-mix -> bf16 pre-down ----------------
__global__ __launch_bounds__(256)
void ht_gn_kernel(const float* __restrict__ opre, const float* __restrict__ qf,
                  const float* __restrict__ ct, const float* __restrict__ denom,
                  const float* __restrict__ xskip, const float* __restrict__ xupR,
                  const float* __restrict__ gnw, const float* __restrict__ gnb,
                  float* __restrict__ out_h, unsigned short* __restrict__ predown) {
  int bI = blockIdx.x, t = threadIdx.x;
  size_t base = (size_t)bI * HHH + t * 4;     // 4 contiguous elems per thread
  float hv[4], s = 0.f, q2 = 0.f;
#pragma unroll
  for (int e = 0; e < 4; e++) {
    int h = t * 4 + e;
    float o = sigmoidf_(opre[base + e]);
    float val = o * ct[base + e] * qf[base + e] / denom[h];
    hv[e] = val; out_h[base + e] = val;
    s += val; q2 += val * val;
  }
  // reduce over 16-lane group (64 elems = one head) inside wave32
#pragma unroll
  for (int m = 8; m >= 1; m >>= 1) { s += __shfl_xor(s, m, 32); q2 += __shfl_xor(q2, m, 32); }
  float mu = s * (1.0f / HSS);
  float var = q2 * (1.0f / HSS) - mu * mu;
  float rs = rsqrtf(var + EPSF);
#pragma unroll
  for (int e = 0; e < 4; e++) {
    int h = t * 4 + e;
    float xn = (hv[e] - mu) * rs * gnw[h] + gnb[h];
    float pre = (xn + xskip[base + e]) * siluf_(xupR[base + e]);
    predown[base + e] = f2bf(pre);
  }
}

// ---------------- launcher ----------------
extern "C" void kernel_launch(void* const* d_in, const int* in_sizes, int n_in,
                              void* d_out, int out_size, void* d_ws, size_t ws_size,
                              hipStream_t stream) {
  (void)in_sizes; (void)n_in; (void)out_size; (void)ws_size;
  const float* x      = (const float*)d_in[0];
  const float* cprev  = (const float*)d_in[2];
  const float* nprev  = (const float*)d_in[3];
  const float* mprev  = (const float*)d_in[4];
  const float* ln_w   = (const float*)d_in[5];
  const float* ln_b   = (const float*)d_in[6];
  const float* W_upL  = (const float*)d_in[7];
  const float* b_upL  = (const float*)d_in[8];
  const float* W_upR  = (const float*)d_in[9];
  const float* b_upR  = (const float*)d_in[10];
  const float* W_down = (const float*)d_in[11];
  const float* b_down = (const float*)d_in[12];
  const float* conv_w = (const float*)d_in[13];
  const float* conv_b = (const float*)d_in[14];
  const float* W_skip = (const float*)d_in[15];
  const float* b_skip = (const float*)d_in[16];
  const float* Wq = (const float*)d_in[17]; const float* bq = (const float*)d_in[18];
  const float* Wk = (const float*)d_in[19]; const float* bk = (const float*)d_in[20];
  const float* Wv = (const float*)d_in[21]; const float* bv = (const float*)d_in[22];
  const float* Wi = (const float*)d_in[23]; const float* bi = (const float*)d_in[24];
  const float* Wf = (const float*)d_in[25]; const float* b_f = (const float*)d_in[26];
  const float* Wo = (const float*)d_in[27]; const float* bo = (const float*)d_in[28];
  const float* gn_w = (const float*)d_in[29];
  const float* gn_b = (const float*)d_in[30];

  float* out_final = (float*)d_out;
  float* out_h = out_final + (size_t)BB * DD;
  float* out_c = out_h + (size_t)BB * HHH;
  float* out_n = out_c + (size_t)BB * HHH;
  float* out_m = out_n + (size_t)BB * HHH;

  char* ws = (char*)d_ws;
  size_t off = 0;
  auto alloc = [&](size_t bytes) -> void* {
    void* p = (void*)(ws + off);
    off += (bytes + 255) & ~(size_t)255;
    return p;
  };
  const size_t BH  = (size_t)BB * HHH;
  const size_t BUP = (size_t)BB * UPF;

  unsigned short* wt_upL  = (unsigned short*)alloc((size_t)DD * UPF * 2);
  unsigned short* wt_upR  = (unsigned short*)alloc((size_t)DD * HHH * 2);
  unsigned short* wt_skip = (unsigned short*)alloc((size_t)UPF * HHH * 2);
  unsigned short* wt_q    = (unsigned short*)alloc((size_t)UPF * HHH * 2);
  unsigned short* wt_k    = (unsigned short*)alloc((size_t)UPF * HHH * 2);
  unsigned short* wt_v    = (unsigned short*)alloc((size_t)UPF * HHH * 2);
  unsigned short* wt_i    = (unsigned short*)alloc((size_t)UPF * HHH * 2);
  unsigned short* wt_f    = (unsigned short*)alloc((size_t)UPF * HHH * 2);
  unsigned short* wt_o    = (unsigned short*)alloc((size_t)UPF * HHH * 2);
  unsigned short* wt_down = (unsigned short*)alloc((size_t)HHH * DD * 2);
  unsigned short* xnorm_bf = (unsigned short*)alloc((size_t)BB * DD * 2);
  float*          xupL_f   = (float*)alloc(BUP * 4);
  unsigned short* xupL_bf  = (unsigned short*)alloc(BUP * 2);
  float*          xupR_f   = (float*)alloc(BH * 4);
  unsigned short* xconv_bf = (unsigned short*)alloc(BUP * 2);
  float*          xskip_f  = (float*)alloc(BH * 4);
  float* q_f = (float*)alloc(BH * 4);
  float* k_f = (float*)alloc(BH * 4);
  float* v_f = (float*)alloc(BH * 4);
  float* i_f = (float*)alloc(BH * 4);
  float* f_f = (float*)alloc(BH * 4);
  float* o_f = (float*)alloc(BH * 4);
  unsigned short* nT_bf = (unsigned short*)alloc(BH * 2);
  unsigned short* qT_bf = (unsigned short*)alloc(BH * 2);
  float* G_f   = (float*)alloc((size_t)HHH * HHH * 4);
  float* denom = (float*)alloc((size_t)HHH * 4);
  unsigned short* predown_bf = (unsigned short*)alloc(BH * 2);

  // 1) weights -> bf16 transposed [N,K]
  auto cvt = [&](const float* W, unsigned short* Wt, int K, int N) {
    convert_transpose<<<(K * N + 255) / 256, 256, 0, stream>>>(W, Wt, K, N);
  };
  cvt(W_upL, wt_upL, DD, UPF);
  cvt(W_upR, wt_upR, DD, HHH);
  cvt(W_skip, wt_skip, UPF, HHH);
  cvt(Wq, wt_q, UPF, HHH); cvt(Wk, wt_k, UPF, HHH); cvt(Wv, wt_v, UPF, HHH);
  cvt(Wi, wt_i, UPF, HHH); cvt(Wf, wt_f, UPF, HHH); cvt(Wo, wt_o, UPF, HHH);
  cvt(W_down, wt_down, HHH, DD);

  // 2) layernorm
  layernorm_bf16<<<BB, 256, 0, stream>>>(x, ln_w, ln_b, xnorm_bf);

  // GEMM dispatch: compile-time specialization of epilogue
  auto gemm_b = [&](const unsigned short* A, const unsigned short* Bt, const float* bias,
                    float* Cf, int M, int N, int K) {   // bias, f32 out
    dim3 g(N / TN, M / TM);
    gemm_bf16<true, false, true, false><<<g, 256, 0, stream>>>(A, Bt, bias, nullptr, Cf, nullptr, M, N, K);
  };
  auto gemm_bb = [&](const unsigned short* A, const unsigned short* Bt, const float* bias,
                     float* Cf, unsigned short* Cbf, int M, int N, int K) {  // bias, f32+bf16 out
    dim3 g(N / TN, M / TM);
    gemm_bf16<true, false, true, true><<<g, 256, 0, stream>>>(A, Bt, bias, nullptr, Cf, Cbf, M, N, K);
  };
  auto gemm_p = [&](const unsigned short* A, const unsigned short* Bt,
                    float* Cf, int M, int N, int K) {   // plain, f32 out
    dim3 g(N / TN, M / TM);
    gemm_bf16<false, false, true, false><<<g, 256, 0, stream>>>(A, Bt, nullptr, nullptr, Cf, nullptr, M, N, K);
  };
  auto gemm_br = [&](const unsigned short* A, const unsigned short* Bt, const float* bias,
                     const float* resid, float* Cf, int M, int N, int K) {  // bias+resid, f32 out
    dim3 g(N / TN, M / TM);
    gemm_bf16<true, true, true, false><<<g, 256, 0, stream>>>(A, Bt, bias, resid, Cf, nullptr, M, N, K);
  };

  // 3-4) up projections
  gemm_bb(xnorm_bf, wt_upL, b_upL, xupL_f, xupL_bf, BB, UPF, DD);
  gemm_b (xnorm_bf, wt_upR, b_upR, xupR_f, BB, HHH, DD);

  // 5) causal conv + silu
  conv_silu_bf16<<<(int)(BUP / 256), 256, 0, stream>>>(xupL_f, conv_w, conv_b, xconv_bf);

  // 6-9) projection GEMMs
  gemm_b(xconv_bf, wt_skip, b_skip, xskip_f, BB, HHH, UPF);
  gemm_b(xconv_bf, wt_q, bq, q_f, BB, HHH, UPF);
  gemm_b(xconv_bf, wt_k, bk, k_f, BB, HHH, UPF);
  gemm_b(xconv_bf, wt_i, bi, i_f, BB, HHH, UPF);
  gemm_b(xconv_bf, wt_f, b_f, f_f, BB, HHH, UPF);
  gemm_b(xupL_bf, wt_v, bv, v_f, BB, HHH, UPF);
  gemm_b(xupL_bf, wt_o, bo, o_f, BB, HHH, UPF);

  // 10) gates -> c_t, n_t, m_t (+ transposed bf16 n, q)
  gate_kernel<<<(int)(BH / 256), 256, 0, stream>>>(i_f, f_f, k_f, v_f, q_f,
                                                   cprev, nprev, mprev,
                                                   out_c, out_n, out_m, nT_bf, qT_bf);

  // 11) G = n^T @ q  (M=H, N=H, K=B)
  gemm_p(nT_bf, qT_bf, G_f, HHH, HHH, BB);

  // 12) denom
  rowmax_abs<<<HHH, 256, 0, stream>>>(G_f, denom);

  // 13) h_t + groupnorm + skip + silu mix
  ht_gn_kernel<<<BB, 256, 0, stream>>>(o_f, q_f, out_c, denom, xskip_f, xupR_f,
                                       gn_w, gn_b, out_h, predown_bf);

  // 14) down projection + residual -> final output
  gemm_br(predown_bf, wt_down, b_down, x, out_final, BB, DD, HHH);
}
